// EgnnBackbone_56135222559303
// MI455X (gfx1250) — matrix operand order repaired and checked by reading, not compiled, not run on hardware
//
#include <hip/hip_runtime.h>
#include <hip/hip_bf16.h>

typedef __bf16 bf16_t;
typedef __attribute__((ext_vector_type(16))) __bf16 v16bf;
typedef __attribute__((ext_vector_type(8)))  __bf16 v8bf;
typedef __attribute__((ext_vector_type(8)))  float  v8f;

#define NNODES  20000
#define NEDGES  512000
#define HDIM    128
#define NLAYERS 2
#define KEDGE   288   // 2*H + 6 = 262, zero-padded to multiple of 32
#define ETILE   64    // edges (rows) per block

__device__ __forceinline__ float silu_f(float v) {
  return v / (1.0f + __expf(-v));
}

#define WMMA_BF16(A_, B_, C_) \
  __builtin_amdgcn_wmma_f32_16x16x32_bf16(false, (A_), false, (B_), (short)0, (C_), false, false)

// A fragment: row-major [16 x K] tile at p (K-contiguous), leading dim ld (multiple of 16).
// ISA 16-bit A 16x32 layout: lanes 0-15 hold K={0..7,16..23}, lanes 16-31 hold K={8..15,24..31}.
__device__ __forceinline__ v16bf load_frag_a(const bf16_t* p, int ld) {
  const int lane = threadIdx.x & 31;
  const bf16_t* row = p + (lane & 15) * ld + ((lane >> 4) << 3);
  v8bf lo = *(const v8bf*)(row);        // K 0..7 (rel)
  v8bf hi = *(const v8bf*)(row + 16);   // K 16..23 (rel)
  v16bf f;
#pragma unroll
  for (int i = 0; i < 8; ++i) { f[i] = lo[i]; f[i + 8] = hi[i]; }
  return f;
}

// Same as above but source is f32 (LDS), converted to bf16 on load.
__device__ __forceinline__ v16bf load_frag_a_f32(const float* p, int ld) {
  const int lane = threadIdx.x & 31;
  const float* row = p + (lane & 15) * ld + ((lane >> 4) << 3);
  v16bf f;
#pragma unroll
  for (int i = 0; i < 8; ++i) {
    f[i]     = (bf16_t)row[i];
    f[i + 8] = (bf16_t)row[i + 16];
  }
  return f;
}

// B fragment for W row-major [128 x K]: B[k][n] = W[n0+n][k0+k].
// ISA B 32x16 layout: lane = column n, lanes 0-15 hold K={0..7,16..23}, lanes 16-31 hold K={8..15,24..31}.
__device__ __forceinline__ v16bf load_frag_b(const bf16_t* p, int ld) {
  const int lane = threadIdx.x & 31;
  const bf16_t* row = p + (lane & 15) * ld + ((lane >> 4) << 3);
  v8bf lo = *(const v8bf*)(row);
  v8bf hi = *(const v8bf*)(row + 16);
  v16bf f;
#pragma unroll
  for (int i = 0; i < 8; ++i) { f[i] = lo[i]; f[i + 8] = hi[i]; }
  return f;
}

// ---------------------------------------------------------------------------
// 128-wide GEMM: Out[M,128] = act(A[M,K] @ W[128,K]^T + bias) [+resid]
// Block = 256 threads (8 waves). 64-row M-tile staged in LDS; each wave owns
// one 16-col tile and accumulates 4 M-subtiles, so every B fragment feeds
// 4 WMMAs (4x reuse of the weight-stream).
// ---------------------------------------------------------------------------
template<int K, bool SILU, bool RESID, bool OUTBF>
__global__ __launch_bounds__(256) void gemm128_kernel(
    const bf16_t* __restrict__ A, int lda,
    const bf16_t* __restrict__ W, int ldw,
    const float* __restrict__ bias,
    float* __restrict__ outf, bf16_t* __restrict__ outb, int ldo, int M)
{
  __shared__ __align__(32) bf16_t As[64 * K];
  const int m0  = blockIdx.x * 64;
  const int tid = threadIdx.x;

  // stage A tile (vectorized 8 x bf16 = 16B chunks), clamp ragged tail rows
#pragma unroll
  for (int idx = tid; idx < 64 * K / 8; idx += 256) {
    const int r  = idx / (K / 8);
    const int c8 = idx - r * (K / 8);
    int row = m0 + r; if (row >= M) row = M - 1;
    *(v8bf*)(As + r * K + c8 * 8) = *(const v8bf*)(A + (size_t)row * lda + c8 * 8);
  }
  __syncthreads();

  const int wave = tid >> 5, lane = tid & 31;
  v8f acc[4] = {};
  const bf16_t* Wn = W + (wave * 16) * ldw;
#pragma unroll
  for (int k0 = 0; k0 < K; k0 += 32) {
    const v16bf b = load_frag_b(Wn + k0, ldw);
#pragma unroll
    for (int mt = 0; mt < 4; ++mt) {
      const v16bf a = load_frag_a(As + mt * 16 * K + k0, K);
      acc[mt] = WMMA_BF16(a, b, acc[mt]);
    }
  }

  const int n  = wave * 16 + (lane & 15);
  const int mo = (lane >> 4) << 3;     // C/D: lanes>=16 hold M = r + 8
  const float bn = bias[n];
#pragma unroll
  for (int mt = 0; mt < 4; ++mt) {
#pragma unroll
    for (int r = 0; r < 8; ++r) {
      float v = acc[mt][r] + bn;
      if (SILU) v = silu_f(v);
      const int row = m0 + mt * 16 + mo + r;
      if (row < M) {
        if (RESID)      outf[(size_t)row * ldo + n] += v;
        else if (OUTBF) outb[(size_t)row * ldo + n] = (bf16_t)v;
        else            outf[(size_t)row * ldo + n] = v;
      }
    }
  }
}

// ---------------------------------------------------------------------------
// Fused edge kernel: build edge features, msg MLP (2 layers), coef MLP,
// segment-sum atomics for m_i and coordinate deltas. 64 edges per block,
// 4 M-subtiles per wave per GEMM (B-fragment reuse x4).
// `ed` (bf16 edge features, dead after GEMM1) and `ts` (f32 coef hidden,
// born in GEMM3) share one LDS union buffer.
// ---------------------------------------------------------------------------
__global__ __launch_bounds__(256) void edge_kernel(
    const float* __restrict__ hp,   const float* __restrict__ xp,
    const int*   __restrict__ srcp, const int*   __restrict__ dstp,
    const float* __restrict__ bpp,  const float* __restrict__ msab,
    const float* __restrict__ chem, const float* __restrict__ roff,
    const float* __restrict__ cbm,
    const bf16_t* __restrict__ w1,  const float* __restrict__ b1,
    const bf16_t* __restrict__ w2,  const float* __restrict__ b2,
    const bf16_t* __restrict__ cw1, const float* __restrict__ cb1,
    const float* __restrict__ cw2,  const float* __restrict__ cb2,
    float* __restrict__ m_acc, float* __restrict__ x_acc)
{
  __shared__ __align__(32) char uni[ETILE * KEDGE * 2]; // 36 KB: ed bf16[64][288] / ts f32[64][128]
  __shared__ __align__(32) bf16_t m1s[ETILE * HDIM];    // 16 KB
  __shared__ __align__(32) float  ms[ETILE * HDIM];     // 32 KB
  __shared__ float diffs[ETILE][3];
  __shared__ float d2s[ETILE];
  __shared__ float coefs[ETILE];
  __shared__ int   srcs[ETILE];
  __shared__ int   dsts[ETILE];
  bf16_t* ed = (bf16_t*)uni;
  float*  ts = (float*)uni;

  const int e0  = blockIdx.x * ETILE;
  const int tid = threadIdx.x;

  if (tid < ETILE) {
    const int e = e0 + tid;
    const int s = srcp[e], d = dstp[e];
    srcs[tid] = s; dsts[tid] = d;
    const float dx = xp[s * 3 + 0] - xp[d * 3 + 0];
    const float dy = xp[s * 3 + 1] - xp[d * 3 + 1];
    const float dz = xp[s * 3 + 2] - xp[d * 3 + 2];
    diffs[tid][0] = dx; diffs[tid][1] = dy; diffs[tid][2] = dz;
    d2s[tid] = dx * dx + dy * dy + dz * dz;
  }
  __syncthreads();

  // stage [h_src | h_dst] columns 0..255 as float4 gathers -> bf16
  for (int idx = tid; idx < ETILE * 64; idx += 256) {
    const int r = idx >> 6;
    const int c = (idx & 63) << 2;
    const float* sp = (c < 128) ? (hp + (size_t)srcs[r] * HDIM + c)
                                : (hp + (size_t)dsts[r] * HDIM + (c - 128));
    const float4 v = *(const float4*)sp;
    bf16_t* o = ed + r * KEDGE + c;
    o[0] = (bf16_t)v.x; o[1] = (bf16_t)v.y; o[2] = (bf16_t)v.z; o[3] = (bf16_t)v.w;
  }
  // stage [dist2 | 5 biases | zero-pad] columns 256..287
  for (int idx = tid; idx < ETILE * 32; idx += 256) {
    const int r = idx >> 5, c = 256 + (idx & 31);
    const int e = e0 + r;
    float v;
    if      (c == 256) v = d2s[r];
    else if (c == 257) v = bpp[e];
    else if (c == 258) v = msab[e];
    else if (c == 259) v = chem[e];
    else if (c == 260) v = roff[e];
    else if (c == 261) v = cbm[e];
    else               v = 0.0f;
    ed[r * KEDGE + c] = (bf16_t)v;
  }
  __syncthreads();

  const int wave = tid >> 5, lane = tid & 31;
  const int n  = wave * 16 + (lane & 15);
  const int mo = (lane >> 4) << 3;

  { // GEMM1: silu(edge @ w1^T + b1) -> m1s (bf16)
    v8f acc[4] = {};
    const bf16_t* Wn = w1 + (wave * 16) * KEDGE;
#pragma unroll
    for (int k0 = 0; k0 < KEDGE; k0 += 32) {
      const v16bf b = load_frag_b(Wn + k0, KEDGE);
#pragma unroll
      for (int mt = 0; mt < 4; ++mt) {
        const v16bf a = load_frag_a(ed + mt * 16 * KEDGE + k0, KEDGE);
        acc[mt] = WMMA_BF16(a, b, acc[mt]);
      }
    }
    const float bn = b1[n];
#pragma unroll
    for (int mt = 0; mt < 4; ++mt)
#pragma unroll
      for (int r = 0; r < 8; ++r)
        m1s[(mt * 16 + mo + r) * HDIM + n] = (bf16_t)silu_f(acc[mt][r] + bn);
  }
  __syncthreads();

  { // GEMM2: silu(m1 @ w2^T + b2) -> ms (f32, full precision for atomics)
    v8f acc[4] = {};
    const bf16_t* Wn = w2 + (wave * 16) * HDIM;
#pragma unroll
    for (int k0 = 0; k0 < HDIM; k0 += 32) {
      const v16bf b = load_frag_b(Wn + k0, HDIM);
#pragma unroll
      for (int mt = 0; mt < 4; ++mt) {
        const v16bf a = load_frag_a(m1s + mt * 16 * HDIM + k0, HDIM);
        acc[mt] = WMMA_BF16(a, b, acc[mt]);
      }
    }
    const float bn = b2[n];
#pragma unroll
    for (int mt = 0; mt < 4; ++mt)
#pragma unroll
      for (int r = 0; r < 8; ++r)
        ms[(mt * 16 + mo + r) * HDIM + n] = silu_f(acc[mt][r] + bn);
  }
  __syncthreads();

  { // GEMM3: silu(m @ cw1^T + cb1) -> ts (f32, aliases dead `ed`)
    v8f acc[4] = {};
    const bf16_t* Wn = cw1 + (wave * 16) * HDIM;
#pragma unroll
    for (int k0 = 0; k0 < HDIM; k0 += 32) {
      const v16bf b = load_frag_b(Wn + k0, HDIM);
#pragma unroll
      for (int mt = 0; mt < 4; ++mt) {
        const v16bf a = load_frag_a_f32(ms + mt * 16 * HDIM + k0, HDIM);
        acc[mt] = WMMA_BF16(a, b, acc[mt]);
      }
    }
    const float bn = cb1[n];
#pragma unroll
    for (int mt = 0; mt < 4; ++mt)
#pragma unroll
      for (int r = 0; r < 8; ++r)
        ts[(mt * 16 + mo + r) * HDIM + n] = silu_f(acc[mt][r] + bn);
  }
  __syncthreads();

  if (tid < ETILE) { // coef = tanh(t . cw2 + cb2)
    float dot = 0.0f;
#pragma unroll 8
    for (int k = 0; k < HDIM; ++k) dot += ts[tid * HDIM + k] * cw2[k];
    coefs[tid] = tanhf(dot + cb2[0]);
  }
  __syncthreads();

  // segment-sum atomics
  for (int idx = tid; idx < ETILE * HDIM; idx += 256) {
    const int r = idx >> 7, c = idx & 127;
    atomicAdd(&m_acc[(size_t)srcs[r] * HDIM + c], ms[idx]);
  }
  if (tid < ETILE * 3) {
    const int r = tid / 3, dco = tid - r * 3;
    atomicAdd(&x_acc[(size_t)srcs[r] * 3 + dco], diffs[r][dco] * coefs[r]);
  }
}

// --------------------------- small helper kernels ---------------------------
__global__ void zero_f32(float* p, int nelem) {
  const int i = blockIdx.x * blockDim.x + threadIdx.x;
  if (i < nelem) p[i] = 0.0f;
}
__global__ void copy_f32(float* dstp, const float* srcp, int nelem) {
  const int i = blockIdx.x * blockDim.x + threadIdx.x;
  if (i < nelem) dstp[i] = srcp[i];
}
__global__ void cvt_pad_bf16(bf16_t* dstp, const float* srcp, int Kin, int Kout, int total) {
  const int i = blockIdx.x * blockDim.x + threadIdx.x;
  if (i >= total) return;
  const int r = i / Kout, c = i - r * Kout;
  dstp[i] = (bf16_t)(c < Kin ? srcp[r * Kin + c] : 0.0f);
}
__global__ void edge_count_kernel(const int* __restrict__ srcp, float* cnt) {
  const int e = blockIdx.x * blockDim.x + threadIdx.x;
  if (e < NEDGES) atomicAdd(&cnt[srcp[e]], 1.0f);
}
__global__ void inv_denom_kernel(float* cnt, int nelem) {
  const int i = blockIdx.x * blockDim.x + threadIdx.x;
  if (i < nelem) cnt[i] = 1.0f / fmaxf(cnt[i], 1.0f);
}
// hcat[n, 0:128]=bf16(h), hcat[n,128:256]=bf16(m_acc*invd); x += x_acc*invd
__global__ void build_hcat_kernel(const float* __restrict__ hp,
                                  const float* __restrict__ macc,
                                  const float* __restrict__ invdp,
                                  const float* __restrict__ xaccp,
                                  float* __restrict__ xp,
                                  bf16_t* __restrict__ hcatp) {
  const int i = blockIdx.x * blockDim.x + threadIdx.x;
  if (i < NNODES * HDIM) {
    const int node = i >> 7, c = i & 127;
    hcatp[node * 256 + c]       = (bf16_t)hp[i];
    hcatp[node * 256 + 128 + c] = (bf16_t)(macc[i] * invdp[node]);
  }
  if (i < NNODES * 3) xp[i] += xaccp[i] * invdp[i / 3];
}
__global__ __launch_bounds__(128) void ln_kernel(
    const float* __restrict__ hp, const float* __restrict__ g,
    const float* __restrict__ bb, float* __restrict__ outp) {
  __shared__ float red[128];
  const int node = blockIdx.x, t = threadIdx.x;
  const float v = hp[node * HDIM + t];
  red[t] = v;
  __syncthreads();
  for (int s = 64; s > 0; s >>= 1) { if (t < s) red[t] += red[t + s]; __syncthreads(); }
  const float mu = red[0] * (1.0f / 128.0f);
  __syncthreads();
  const float dv = v - mu;
  red[t] = dv * dv;
  __syncthreads();
  for (int s = 64; s > 0; s >>= 1) { if (t < s) red[t] += red[t + s]; __syncthreads(); }
  const float var = red[0] * (1.0f / 128.0f);
  outp[node * HDIM + t] = dv * rsqrtf(var + 1e-5f) * g[t] + bb[t];
}

// ---------------------------------------------------------------------------
extern "C" void kernel_launch(void* const* d_in, const int* in_sizes, int n_in,
                              void* d_out, int out_size, void* d_ws, size_t ws_size,
                              hipStream_t stream) {
  (void)in_sizes; (void)n_in; (void)out_size; (void)ws_size;
  const float* node_features = (const float*)d_in[0];
  const float* coords  = (const float*)d_in[1];
  const int*   src     = (const int*)d_in[2];
  const int*   dst     = (const int*)d_in[3];
  const float* bpp     = (const float*)d_in[4];
  const float* msab    = (const float*)d_in[5];
  const float* chem    = (const float*)d_in[6];
  const float* roff    = (const float*)d_in[7];
  const float* cbm     = (const float*)d_in[8];
  const float* proj_w  = (const float*)d_in[9];
  const float* proj_b  = (const float*)d_in[10];
  const float* msg_w1  = (const float*)d_in[11];
  const float* msg_b1  = (const float*)d_in[12];
  const float* msg_w2  = (const float*)d_in[13];
  const float* msg_b2  = (const float*)d_in[14];
  const float* hm_w1   = (const float*)d_in[15];
  const float* hm_b1   = (const float*)d_in[16];
  const float* hm_w2   = (const float*)d_in[17];
  const float* hm_b2   = (const float*)d_in[18];
  const float* co_w1   = (const float*)d_in[19];
  const float* co_b1   = (const float*)d_in[20];
  const float* co_w2   = (const float*)d_in[21];
  const float* co_b2   = (const float*)d_in[22];
  const float* ln_g    = (const float*)d_in[23];
  const float* ln_b    = (const float*)d_in[24];
  float* out = (float*)d_out;

  // workspace bump allocator (256B aligned)
  char* wsp = (char*)d_ws;
  auto alloc = [&](size_t bytes) -> char* {
    char* p = wsp; wsp += ((bytes + 255) & ~(size_t)255); return p;
  };
  float*  h     = (float*)alloc((size_t)NNODES * HDIM * 4);
  bf16_t* hcat  = (bf16_t*)alloc((size_t)NNODES * 256 * 2);
  float*  xcur  = (float*)alloc((size_t)NNODES * 3 * 4);
  float*  m_acc = (float*)alloc((size_t)NNODES * HDIM * 4);
  float*  x_acc = (float*)alloc((size_t)NNODES * 3 * 4);
  float*  invd  = (float*)alloc((size_t)NNODES * 4);
  bf16_t* tmpbf = (bf16_t*)alloc((size_t)NNODES * HDIM * 2); // nf_bf, later hu_bf
  bf16_t* wproj = (bf16_t*)alloc((size_t)128 * 128 * 2);
  bf16_t* wmsg1 = (bf16_t*)alloc((size_t)NLAYERS * 128 * KEDGE * 2);
  bf16_t* wmsg2 = (bf16_t*)alloc((size_t)NLAYERS * 128 * 128 * 2);
  bf16_t* whm1  = (bf16_t*)alloc((size_t)NLAYERS * 128 * 256 * 2);
  bf16_t* whm2  = (bf16_t*)alloc((size_t)NLAYERS * 128 * 128 * 2);
  bf16_t* wco1  = (bf16_t*)alloc((size_t)NLAYERS * 128 * 128 * 2);

  const int T = 256;
  auto blk = [](int nelem, int t) { return (nelem + t - 1) / t; };
  const int gemmGrid = (NNODES + 63) / 64;

  // weight / input conversions to bf16
  cvt_pad_bf16<<<blk(128 * 128, T), T, 0, stream>>>(wproj, proj_w, 128, 128, 128 * 128);
  cvt_pad_bf16<<<blk(NLAYERS * 128 * KEDGE, T), T, 0, stream>>>(wmsg1, msg_w1, 262, KEDGE, NLAYERS * 128 * KEDGE);
  cvt_pad_bf16<<<blk(NLAYERS * 128 * 128, T), T, 0, stream>>>(wmsg2, msg_w2, 128, 128, NLAYERS * 128 * 128);
  cvt_pad_bf16<<<blk(NLAYERS * 128 * 256, T), T, 0, stream>>>(whm1, hm_w1, 256, 256, NLAYERS * 128 * 256);
  cvt_pad_bf16<<<blk(NLAYERS * 128 * 128, T), T, 0, stream>>>(whm2, hm_w2, 128, 128, NLAYERS * 128 * 128);
  cvt_pad_bf16<<<blk(NLAYERS * 128 * 128, T), T, 0, stream>>>(wco1, co_w1, 128, 128, NLAYERS * 128 * 128);
  cvt_pad_bf16<<<blk(NNODES * 128, T), T, 0, stream>>>(tmpbf, node_features, 128, 128, NNODES * 128);

  // x init, degree counts
  copy_f32<<<blk(NNODES * 3, T), T, 0, stream>>>(xcur, coords, NNODES * 3);
  zero_f32<<<blk(NNODES, T), T, 0, stream>>>(invd, NNODES);
  edge_count_kernel<<<blk(NEDGES, T), T, 0, stream>>>(src, invd);
  inv_denom_kernel<<<blk(NNODES, T), T, 0, stream>>>(invd, NNODES);

  // h = node_features @ proj_w^T + proj_b
  gemm128_kernel<128, false, false, false><<<gemmGrid, 256, 0, stream>>>(
      tmpbf, 128, wproj, 128, proj_b, h, nullptr, 128, NNODES);

  for (int l = 0; l < NLAYERS; ++l) {
    zero_f32<<<blk(NNODES * HDIM, T), T, 0, stream>>>(m_acc, NNODES * HDIM);
    zero_f32<<<blk(NNODES * 3, T), T, 0, stream>>>(x_acc, NNODES * 3);

    edge_kernel<<<NEDGES / ETILE, 256, 0, stream>>>(
        h, xcur, src, dst, bpp, msab, chem, roff, cbm,
        wmsg1 + (size_t)l * 128 * KEDGE, msg_b1 + l * 128,
        wmsg2 + (size_t)l * 128 * 128,   msg_b2 + l * 128,
        wco1  + (size_t)l * 128 * 128,   co_b1  + l * 128,
        co_w2 + l * 128,                 co_b2  + l,
        m_acc, x_acc);

    build_hcat_kernel<<<blk(NNODES * HDIM, T), T, 0, stream>>>(
        h, m_acc, invd, x_acc, xcur, hcat);

    // hu = silu([h, m_i] @ hm_w1^T + b1)   (bf16 out for next GEMM)
    gemm128_kernel<256, true, false, true><<<gemmGrid, 256, 0, stream>>>(
        hcat, 256, whm1 + (size_t)l * 128 * 256, 256, hm_b1 + l * 128,
        nullptr, tmpbf, 128, NNODES);
    // h += hu @ hm_w2^T + b2
    gemm128_kernel<128, false, true, false><<<gemmGrid, 256, 0, stream>>>(
        tmpbf, 128, whm2 + (size_t)l * 128 * 128, 128, hm_b2 + l * 128,
        h, nullptr, 128, NNODES);
  }

  // output: [h_out (LayerNorm) | x]
  ln_kernel<<<NNODES, 128, 0, stream>>>(h, ln_g, ln_b, out);
  copy_f32<<<blk(NNODES * 3, T), T, 0, stream>>>(out + (size_t)NNODES * HDIM, xcur, NNODES * 3);
}
